// LinearBN1d_if_83322365542498
// MI455X (gfx1250) — matrix operand order repaired and checked by reading, not compiled, not run on hardware
//
#include <hip/hip_runtime.h>

// ---------------------------------------------------------------------------
// SNN Linear+BN-fold + integrate-and-fire, tuned for MI455X (gfx1250, wave32)
//   pot = x_st(f16) @ w_norm^T(f16)  via v_wmma_f32_16x16x32_f16
//   - weight tiles: GLOBAL_LOAD_ASYNC_TO_LDS_B128 (GVS form, ASYNCcnt),
//     double-buffered in LDS so the copy overlaps the WMMA stream
//   - spike tiles: f32->f16 via VALU + ds_store_b128, double-buffered
//   - IF scan fused (block owns all T=32 rows of one sample; pot stays in LDS)
// ---------------------------------------------------------------------------

typedef __attribute__((ext_vector_type(16))) _Float16 v16h;
typedef __attribute__((ext_vector_type(8)))  _Float16 v8h;
typedef __attribute__((ext_vector_type(4)))  _Float16 v4h;
typedef __attribute__((ext_vector_type(8)))  float    v8f;
typedef __attribute__((ext_vector_type(4)))  float    v4f;

#define D_IN   2048
#define D_OUT  2048
#define NSAMP  256
#define TSTEPS 32
#define KC     64      // K-chunk staged in LDS (halves)
#define BN_T   256     // output columns per block
#define RVEC   9       // padded row stride in v8h units (72 halves = 36 dwords)

// ---------------------------------------------------------------------------
// Fold BN into weights:  ratio = gamma / sqrt(var)   (reference: NO eps here)
// ---------------------------------------------------------------------------
__global__ __launch_bounds__(256)
void fold_bn_kernel(const float* __restrict__ w, const float* __restrict__ b,
                    const float* __restrict__ gamma, const float* __restrict__ beta,
                    const float* __restrict__ rmean, const float* __restrict__ rvar,
                    _Float16* __restrict__ wh, float* __restrict__ bnorm) {
    const int o   = blockIdx.x;
    const int tid = threadIdx.x;
    const float ratio = gamma[o] / sqrtf(rvar[o]);
    const float* wr = w  + (size_t)o * D_IN;
    _Float16*   whr = wh + (size_t)o * D_IN;
    const int i0 = tid * 8;
    v4f f0 = *(const v4f*)(wr + i0);
    v4f f1 = *(const v4f*)(wr + i0 + 4);
    f0 = f0 * ratio;
    f1 = f1 * ratio;
    v4h h0 = __builtin_convertvector(f0, v4h);
    v4h h1 = __builtin_convertvector(f1, v4h);
    *(v8h*)&whr[i0] = __builtin_shufflevector(h0, h1, 0,1,2,3,4,5,6,7);
    if (tid == 0)
        bnorm[o] = (b[o] - rmean[o]) * ratio + beta[o];
}

// ---------------------------------------------------------------------------
struct Stage {
    v8h xs[TSTEPS][RVEC];   // A staging: 32 rows x 72 halves (64 + pad)
    v8h ws[BN_T][RVEC];     // B staging: 256 cols x 72 halves (64 + pad)
};

// issue 8 async global->LDS b128 copies (GVS: uniform SGPR base + 32-bit voff)
__device__ __forceinline__ void stage_w_async(const _Float16* whk,
                                              const unsigned* ldsdst,
                                              const unsigned* voff) {
    #pragma unroll
    for (int r = 0; r < 8; ++r)
        asm volatile("global_load_async_to_lds_b128 %0, %1, %2"
                     :: "v"(ldsdst[r]), "v"(voff[r]), "s"(whk)
                     : "memory");
}

// convert one v8h worth of spikes f32->f16 and store to LDS
__device__ __forceinline__ void stage_x(Stage* sb, const float* gx,
                                        int row, int kk) {
    v4f f0 = *(const v4f*)gx;
    v4f f1 = *(const v4f*)(gx + 4);
    v4h h0 = __builtin_convertvector(f0, v4h);
    v4h h1 = __builtin_convertvector(f1, v4h);
    sb->xs[row][kk] = __builtin_shufflevector(h0, h1, 0,1,2,3,4,5,6,7);
}

// 2 K-steps of 32 -> 8 WMMAs per wave per chunk
__device__ __forceinline__ void compute_chunk(const Stage* sb, v8f (&acc)[2][2],
                                              int wave, int sel, int l15) {
    #pragma unroll
    for (int ks = 0; ks < 2; ++ks) {
        v16h a[2], bm[2];
        #pragma unroll
        for (int mi = 0; mi < 2; ++mi) {
            // A (16-bit 16x32): lane m=l15 holds K runs {sel*8..} U {16+sel*8..}
            const int row = mi * 16 + l15;
            v8h lo = sb->xs[row][ks * 4 + sel];
            v8h hi = sb->xs[row][ks * 4 + 2 + sel];
            a[mi] = __builtin_shufflevector(lo, hi,
                    0,1,2,3,4,5,6,7,8,9,10,11,12,13,14,15);
        }
        #pragma unroll
        for (int ni = 0; ni < 2; ++ni) {
            // B (16-bit 32x16): lane n=l15 holds K runs {sel*8..} U {16+sel*8..}
            const int c = wave * 32 + ni * 16 + l15;
            v8h lo = sb->ws[c][ks * 4 + sel];
            v8h hi = sb->ws[c][ks * 4 + 2 + sel];
            bm[ni] = __builtin_shufflevector(lo, hi,
                    0,1,2,3,4,5,6,7,8,9,10,11,12,13,14,15);
        }
        #pragma unroll
        for (int mi = 0; mi < 2; ++mi)
            #pragma unroll
            for (int ni = 0; ni < 2; ++ni)
                acc[mi][ni] = __builtin_amdgcn_wmma_f32_16x16x32_f16(
                    /*neg_a=*/false, a[mi], /*neg_b=*/false, bm[ni],
                    /*c_mod=*/(short)0, acc[mi][ni],
                    /*reuse_a=*/false, /*reuse_b=*/false);
    }
}

// ---------------------------------------------------------------------------
// Main kernel: per block -> one sample n (M = 32 timestep rows) x 256 columns.
// 8 waves; wave w owns columns [w*32, w*32+32): 2 M-tiles x 2 N-tiles of 16x16.
// Double-buffered K pipeline: wait(i) -> barrier -> issue(i+1) -> compute(i).
// ---------------------------------------------------------------------------
__global__ __launch_bounds__(256)
void snn_gemm_if_kernel(const float*    __restrict__ x,      // [N, T, D_IN] f32 (0/1)
                        const _Float16* __restrict__ wh,     // [D_OUT, D_IN] f16 folded
                        const float*    __restrict__ bnorm,  // [D_OUT]
                        float*          __restrict__ spike_out,  // [N, T, D_OUT]
                        float*          __restrict__ spike_cnt)  // [N, D_OUT]
{
    __shared__ union {
        Stage stg[2];               // 2 x 41.5 KB staging buffers
        float pot[TSTEPS][BN_T];    // epilogue potentials (reuses same LDS)
    } S;

    const int tid  = threadIdx.x;
    const int lane = tid & 31;
    const int wave = tid >> 5;
    const int n    = blockIdx.x;
    const int col0 = blockIdx.y * BN_T;

    const int sel = lane >> 4;   // half-wave select (A/B K-run placement)
    const int l15 = lane & 15;

    v8f acc[2][2] = {};

    // ---- loop-invariant async-copy addressing (computed once) ----
    unsigned voff[8], dst0[8], dst1[8];
    {
        const int sub   = tid & 7;
        const int rbase = tid >> 3;
        #pragma unroll
        for (int r = 0; r < 8; ++r) {
            const int c = rbase + r * 32;
            voff[r] = ((unsigned)(col0 + c) * (unsigned)D_IN
                       + (unsigned)(sub * 8)) * 2u;               // global byte offset
            dst0[r] = (unsigned)(uintptr_t)&S.stg[0].ws[c][sub];  // LDS byte address
            dst1[r] = (unsigned)(uintptr_t)&S.stg[1].ws[c][sub];
        }
    }
    const int    xrow = tid >> 3;
    const int    xkk  = tid & 7;
    const float* gx   = x + ((size_t)n * TSTEPS + xrow) * D_IN + xkk * 8;

    // ---- prologue: stage chunk 0 into buffer 0 ----
    stage_w_async(wh, dst0, voff);
    stage_x(&S.stg[0], gx, xrow, xkk);

    for (int k0 = 0; k0 < D_IN; k0 += 2 * KC) {
        // ===== body A: compute chunk k0 (buf0), stage k0+KC into buf1 =====
        asm volatile("s_wait_asynccnt 0x0" ::: "memory");
        __syncthreads();
        stage_w_async(wh + (k0 + KC), dst1, voff);
        stage_x(&S.stg[1], gx + (k0 + KC), xrow, xkk);
        __builtin_prefetch(gx + (k0 + 2 * KC), 0, 1);   // x, one chunk ahead
        compute_chunk(&S.stg[0], acc, wave, sel, l15);

        // ===== body B: compute chunk k0+KC (buf1), stage k0+2KC into buf0 =====
        asm volatile("s_wait_asynccnt 0x0" ::: "memory");
        __syncthreads();
        if (k0 + 2 * KC < D_IN) {
            stage_w_async(wh + (k0 + 2 * KC), dst0, voff);
            stage_x(&S.stg[0], gx + (k0 + 2 * KC), xrow, xkk);
        }
        compute_chunk(&S.stg[1], acc, wave, sel, l15);
    }

    __syncthreads();   // staging LDS no longer needed -> reuse as pot[][]

    // ---- dump accumulators: C layout (VGPR r: lanes<16 -> M=r, lanes>=16 -> M=r+8)
    {
        const int lrow = sel * 8;
        #pragma unroll
        for (int mi = 0; mi < 2; ++mi)
            #pragma unroll
            for (int ni = 0; ni < 2; ++ni) {
                const int c = wave * 32 + ni * 16 + l15;
                #pragma unroll
                for (int r = 0; r < 8; ++r)
                    S.pot[mi * 16 + lrow + r][c] = acc[mi][ni][r];
            }
    }
    __syncthreads();

    // ---- integrate-and-fire scan: one column per thread (conflict-free reads)
    {
        const int o   = col0 + tid;
        const float bn = bnorm[o];
        float v = 0.0f, cnt = 0.0f;
        float* so = spike_out + ((size_t)n * TSTEPS) * D_OUT + o;
        #pragma unroll
        for (int t = 0; t < TSTEPS; ++t) {
            v += S.pot[t][tid] + bn;
            const float s = (v >= 1.0f) ? 1.0f : 0.0f;   // VTH = 1.0
            v -= s;                                       // subtractive reset
            so[(size_t)t * D_OUT] = s;
            cnt += s;
        }
        spike_cnt[(size_t)n * D_OUT + o] = cnt;
    }
}

// ---------------------------------------------------------------------------
extern "C" void kernel_launch(void* const* d_in, const int* in_sizes, int n_in,
                              void* d_out, int out_size, void* d_ws, size_t ws_size,
                              hipStream_t stream) {
    const float* x_st  = (const float*)d_in[0];   // [N, T, D_IN]
    // d_in[1] = x_sc : ANN-path input, unused (ANN output not returned)
    const float* w     = (const float*)d_in[2];   // [D_OUT, D_IN]
    const float* b     = (const float*)d_in[3];
    const float* gamma = (const float*)d_in[4];
    const float* beta  = (const float*)d_in[5];
    const float* rmean = (const float*)d_in[6];
    const float* rvar  = (const float*)d_in[7];

    // workspace: folded f16 weights (8 MB) then folded bias (8 KB)
    _Float16* wh    = (_Float16*)d_ws;
    float*    bnorm = (float*)((char*)d_ws + (size_t)D_OUT * D_IN * sizeof(_Float16));

    float* spike_out = (float*)d_out;                                  // [N, T, D_OUT]
    float* spike_cnt = spike_out + (size_t)NSAMP * TSTEPS * D_OUT;     // [N, D_OUT]

    fold_bn_kernel<<<D_OUT, 256, 0, stream>>>(w, b, gamma, beta, rmean, rvar,
                                              wh, bnorm);
    snn_gemm_if_kernel<<<dim3(NSAMP, D_OUT / BN_T), 256, 0, stream>>>(
        x_st, wh, bnorm, spike_out, spike_cnt);
}
